// MLP_14465449853281
// MI455X (gfx1250) — 8-bit matrix operand order repaired and checked by reading, compiled, not run on hardware
//
#include <hip/hip_runtime.h>
#include <hip/hip_bf16.h>
#include <type_traits>

typedef __attribute__((ext_vector_type(8))) int v8i;

#define EPSQ 1e-5f

#if __has_builtin(__builtin_amdgcn_global_load_async_to_lds_b128)
#define USE_ASYNC_LDS 1
#if __has_builtin(__builtin_amdgcn_s_wait_asynccnt)
#define WAIT_ASYNC(n) __builtin_amdgcn_s_wait_asynccnt(n)
#else
#define WAIT_ASYNC(n) asm volatile("s_wait_asynccnt %0" ::"i"(n) : "memory")
#endif
typedef int v4i_vs __attribute__((vector_size(16)));
typedef __attribute__((address_space(1))) v4i_vs* gv4p;
typedef __attribute__((address_space(3))) v4i_vs* lv4p;

__device__ __forceinline__ void async_b128(const int8_t* g, int8_t* l) {
    gv4p gp = (gv4p)(__attribute__((address_space(1))) void*)(int8_t*)g;
    lv4p lp = (lv4p)(__attribute__((address_space(3))) void*)l;
    __builtin_amdgcn_global_load_async_to_lds_b128(gp, lp,  0,  0);
}
#else
#define USE_ASYNC_LDS 0
#endif

__device__ __forceinline__ int clampi(int v, int lo, int hi) {
    return v < lo ? lo : (v > hi ? hi : v);
}

__global__ __launch_bounds__(256)
void abs_sum_k(const float* __restrict__ w, int n4, float* __restrict__ out)
{
    const int tid = threadIdx.x;
    float s = 0.f;
    const float4* w4 = (const float4*)w;
    for (int i = blockIdx.x * 256 + tid; i < n4; i += gridDim.x * 256) {
        float4 v = w4[i];
        s += fabsf(v.x) + fabsf(v.y) + fabsf(v.z) + fabsf(v.w);
    }
    __shared__ float red[256];
    red[tid] = s;
    __syncthreads();
    for (int off = 128; off > 0; off >>= 1) {
        if (tid < off) red[tid] += red[tid + off];
        __syncthreads();
    }
    if (tid == 0) atomicAdd(out, red[0]);
}

__global__ __launch_bounds__(256)
void quant_w_k(const float* __restrict__ w, int8_t* __restrict__ wq,
               int n4, const float* __restrict__ sum, float inv_count)
{
    const float wsc = fmaxf(sum[0] * inv_count, EPSQ);
    const float inv = 1.0f / wsc;
    const int i = blockIdx.x * 256 + threadIdx.x;
    if (i >= n4) return;
    float4 v = ((const float4*)w)[i];
    int q0 = clampi(__float2int_rn(v.x * inv), -1, 1);
    int q1 = clampi(__float2int_rn(v.y * inv), -1, 1);
    int q2 = clampi(__float2int_rn(v.z * inv), -1, 1);
    int q3 = clampi(__float2int_rn(v.w * inv), -1, 1);
    ((int*)wq)[i] = (q0 & 255) | ((q1 & 255) << 8) | ((q2 & 255) << 16) | ((q3 & 255) << 24);
}

__global__ __launch_bounds__(256)
void quant_row_f32(const float* __restrict__ x, int8_t* __restrict__ xq,
                   float* __restrict__ sc, int D)
{
    const int row = blockIdx.x, tid = threadIdx.x;
    float4 v = ((const float4*)(x + (size_t)row * D))[tid];
    float m = fmaxf(fmaxf(fabsf(v.x), fabsf(v.y)), fmaxf(fabsf(v.z), fabsf(v.w)));
    __shared__ float red[256];
    red[tid] = m;
    __syncthreads();
    for (int off = 128; off > 0; off >>= 1) {
        if (tid < off) red[tid] = fmaxf(red[tid], red[tid + off]);
        __syncthreads();
    }
    const float s = fmaxf(red[0], EPSQ);
    const float inv = 127.0f / s;
    int q0 = clampi(__float2int_rn(v.x * inv), -128, 127);
    int q1 = clampi(__float2int_rn(v.y * inv), -128, 127);
    int q2 = clampi(__float2int_rn(v.z * inv), -128, 127);
    int q3 = clampi(__float2int_rn(v.w * inv), -128, 127);
    ((int*)(xq + (size_t)row * D))[tid] =
        (q0 & 255) | ((q1 & 255) << 8) | ((q2 & 255) << 16) | ((q3 & 255) << 24);
    if (tid == 0) sc[row] = s * (1.0f / 127.0f);
}

__global__ __launch_bounds__(256)
void quant_row_bf16(const __hip_bfloat16* __restrict__ h, int8_t* __restrict__ hq,
                    float* __restrict__ sc, int D)
{
    const int row = blockIdx.x, tid = threadIdx.x;
    const int4* src = (const int4*)(h + (size_t)row * D) + tid * 2;
    int4 r0 = src[0];
    int4 r1 = src[1];
    unsigned d[8] = {(unsigned)r0.x, (unsigned)r0.y, (unsigned)r0.z, (unsigned)r0.w,
                     (unsigned)r1.x, (unsigned)r1.y, (unsigned)r1.z, (unsigned)r1.w};
    float v[16];
    float m = 0.f;
#pragma unroll
    for (int j = 0; j < 8; ++j) {
        v[2 * j]     = __uint_as_float(d[j] << 16);
        v[2 * j + 1] = __uint_as_float(d[j] & 0xffff0000u);
        m = fmaxf(m, fmaxf(fabsf(v[2 * j]), fabsf(v[2 * j + 1])));
    }
    __shared__ float red[256];
    red[tid] = m;
    __syncthreads();
    for (int off = 128; off > 0; off >>= 1) {
        if (tid < off) red[tid] = fmaxf(red[tid], red[tid + off]);
        __syncthreads();
    }
    const float s = fmaxf(red[0], EPSQ);
    const float inv = 127.0f / s;
    int pk[4];
#pragma unroll
    for (int j = 0; j < 4; ++j) {
        int q0 = clampi(__float2int_rn(v[4 * j + 0] * inv), -128, 127);
        int q1 = clampi(__float2int_rn(v[4 * j + 1] * inv), -128, 127);
        int q2 = clampi(__float2int_rn(v[4 * j + 2] * inv), -128, 127);
        int q3 = clampi(__float2int_rn(v[4 * j + 3] * inv), -128, 127);
        pk[j] = (q0 & 255) | ((q1 & 255) << 8) | ((q2 & 255) << 16) | ((q3 & 255) << 24);
    }
    int4 outv; outv.x = pk[0]; outv.y = pk[1]; outv.z = pk[2]; outv.w = pk[3];
    ((int4*)(hq + (size_t)row * D))[tid] = outv;
    if (tid == 0) sc[row] = s * (1.0f / 127.0f);
}

template <bool GELU, typename OutT>
__global__ __launch_bounds__(256)
void bitgemm_k(const int8_t* __restrict__ A, const int8_t* __restrict__ Bw,
               const float* __restrict__ a_scale, const float* __restrict__ wsum,
               float winv_count, const float* __restrict__ bias,
               OutT* __restrict__ out, int M, int N, int K)
{
    __shared__ int8_t lA[2][128 * 64];
    __shared__ int8_t lB[2][128 * 64];

    const int tid  = threadIdx.x;
    const int lane = tid & 31;
    const int wave = tid >> 5;
    const int wm   = wave >> 1;
    const int wn   = wave & 1;
    const int lr   = lane & 15;
    const int lh   = lane >> 4;

    const int blockM = blockIdx.y * 128;
    const int blockN = blockIdx.x * 128;

    const int srow = tid >> 1;
    const int sseg = (tid & 1) * 32;

    const int8_t* gA = A  + (size_t)(blockM + srow) * K + sseg;
    const int8_t* gB = Bw + (size_t)(blockN + srow) * K + sseg;
    const int soff = srow * 64 + sseg;

    v8i acc[2][4] = {};
    const int nsteps = K >> 6;

#if USE_ASYNC_LDS
    async_b128(gA +  0, &lA[0][soff]);
    async_b128(gA + 16, &lA[0][soff + 16]);
    async_b128(gB +  0, &lB[0][soff]);
    async_b128(gB + 16, &lB[0][soff + 16]);
#endif

    for (int i = 0; i < nsteps; ++i) {
        const int cur = i & 1;
        const int k0  = i << 6;

#if USE_ASYNC_LDS
        if (i + 1 < nsteps) {
            const int kn = k0 + 64;
            async_b128(gA + kn,      &lA[cur ^ 1][soff]);
            async_b128(gA + kn + 16, &lA[cur ^ 1][soff + 16]);
            async_b128(gB + kn,      &lB[cur ^ 1][soff]);
            async_b128(gB + kn + 16, &lB[cur ^ 1][soff + 16]);
            WAIT_ASYNC(4);
        } else {
            WAIT_ASYNC(0);
        }
        __syncthreads();
#else
        int4 a0 = ((const int4*)(gA + k0))[0];
        int4 a1 = ((const int4*)(gA + k0))[1];
        int4 b0 = ((const int4*)(gB + k0))[0];
        int4 b1 = ((const int4*)(gB + k0))[1];
        *(int4*)(&lA[0][soff])      = a0;
        *(int4*)(&lA[0][soff + 16]) = a1;
        *(int4*)(&lB[0][soff])      = b0;
        *(int4*)(&lB[0][soff + 16]) = b1;
        if (k0 + 64 < K) {
            __builtin_prefetch(gA + k0 + 64, 0, 3);
            __builtin_prefetch(gB + k0 + 64, 0, 3);
        }
        __syncthreads();
#endif

        const int8_t* bufA = USE_ASYNC_LDS ? lA[cur] : lA[0];
        const int8_t* bufB = USE_ASYNC_LDS ? lB[cur] : lB[0];

        v8i af[2];
#pragma unroll
        for (int mi = 0; mi < 2; ++mi) {
            const int8_t* base = bufA + (wm * 32 + mi * 16 + lr) * 64 + lh * 16;
#pragma unroll
            for (int g = 0; g < 4; ++g) {
                uint2 dq = *(const uint2*)(base + (g >> 1) * 32 + (g & 1) * 8);
                af[mi][2 * g]     = (int)dq.x;
                af[mi][2 * g + 1] = (int)dq.y;
            }
        }
        v8i bf[4];
#pragma unroll
        for (int ni = 0; ni < 4; ++ni) {
            const int8_t* base = bufB + (wn * 64 + ni * 16 + lr) * 64 + lh * 16;
            int4 lo = *(const int4*)(base);
            int4 hi = *(const int4*)(base + 32);
            bf[ni][0] = lo.x; bf[ni][1] = lo.y; bf[ni][2] = lo.z; bf[ni][3] = lo.w;
            bf[ni][4] = hi.x; bf[ni][5] = hi.y; bf[ni][6] = hi.z; bf[ni][7] = hi.w;
        }

#pragma unroll
        for (int mi = 0; mi < 2; ++mi)
#pragma unroll
            for (int ni = 0; ni < 4; ++ni)
                acc[mi][ni] = __builtin_amdgcn_wmma_i32_16x16x64_iu8(
                     true, af[mi],  true, bf[ni],
                    acc[mi][ni],  false,  false);

        __syncthreads();
    }

    const float wsc = fmaxf(wsum[0] * winv_count, EPSQ);

#pragma unroll
    for (int mi = 0; mi < 2; ++mi) {
#pragma unroll
        for (int v = 0; v < 8; ++v) {
            const int gm = blockM + wm * 32 + mi * 16 + lh * 8 + v;
            const float rs = a_scale[gm] * wsc;
#pragma unroll
            for (int ni = 0; ni < 4; ++ni) {
                const int gn = blockN + wn * 64 + ni * 16 + lr;
                float y = (float)acc[mi][ni][v] * rs + bias[gn];
                if (GELU) y = 0.5f * y * (1.0f + erff(y * 0.70710678118654752f));
                const size_t idx = (size_t)gm * N + gn;
                if constexpr (std::is_same<OutT, float>::value) out[idx] = y;
                else                                            out[idx] = __float2bfloat16(y);
            }
        }
    }
}

extern "C" void kernel_launch(void* const* d_in, const int* in_sizes, int n_in,
                              void* d_out, int out_size, void* d_ws, size_t ws_size,
                              hipStream_t stream)
{
    const float* x  = (const float*)d_in[0];
    const float* W1 = (const float*)d_in[1];
    const float* b1 = (const float*)d_in[2];
    const float* W2 = (const float*)d_in[3];
    const float* b2 = (const float*)d_in[4];

    const int D = 1024, H = 4096;
    const int T = 8 * 4096;
    const size_t nW1 = (size_t)H * D;
    const size_t nW2 = (size_t)D * H;

    char* ws = (char*)d_ws;
    float*  sums = (float*)ws;
    float*  xsc  = (float*)(ws + 256);
    float*  hsc  = xsc + T;
    int8_t* w1q  = (int8_t*)(ws + 256 + 2 * (size_t)T * 4);
    int8_t* w2q  = w1q + nW1;
    int8_t* xq   = w2q + nW2;
    int8_t* hq   = xq + (size_t)T * D;
    __hip_bfloat16* hbuf = (__hip_bfloat16*)(hq + (size_t)T * H);

    (void)hipMemsetAsync(sums, 0, 256, stream);

    abs_sum_k<<<1024, 256, 0, stream>>>(W1, (int)(nW1 / 4), &sums[0]);
    abs_sum_k<<<1024, 256, 0, stream>>>(W2, (int)(nW2 / 4), &sums[1]);

    quant_w_k<<<(int)(nW1 / 4 / 256), 256, 0, stream>>>(W1, w1q, (int)(nW1 / 4),
                                                        &sums[0], 1.0f / (float)nW1);
    quant_w_k<<<(int)(nW2 / 4 / 256), 256, 0, stream>>>(W2, w2q, (int)(nW2 / 4),
                                                        &sums[1], 1.0f / (float)nW2);

    quant_row_f32<<<T, 256, 0, stream>>>(x, xq, xsc, D);

    dim3 g1(H / 128, T / 128);
    bitgemm_k<true, __hip_bfloat16><<<g1, 256, 0, stream>>>(
        xq, w1q, xsc, &sums[0], 1.0f / (float)nW1, b1, hbuf, T, H, D);

    quant_row_bf16<<<T, 256, 0, stream>>>(hbuf, hq, hsc, H);

    dim3 g2(D / 128, T / 128);
    bitgemm_k<false, float><<<g2, 256, 0, stream>>>(
        hq, w2q, hsc, &sums[1], 1.0f / (float)nW2, b2, (float*)d_out, T, D, H);
}
